// VQVae_70394513981932
// MI455X (gfx1250) — matrix operand order repaired and checked by reading, not compiled, not run on hardware
//
#include <hip/hip_runtime.h>
#include <math.h>

// ---------------------------------------------------------------------------
// Types for CDNA5 WMMA (wave32, 16x16x32 bf16 -> f32 accumulate)
// ---------------------------------------------------------------------------
typedef __attribute__((ext_vector_type(16))) __bf16          v16bf;
typedef __attribute__((ext_vector_type(16))) unsigned short  v16u;
typedef __attribute__((ext_vector_type(8)))  float           v8f;

union V16Cast { v16u u; v16bf b; };

__device__ __forceinline__ float bf2f(unsigned short u) {
  unsigned int x = ((unsigned int)u) << 16;
  float f; __builtin_memcpy(&f, &x, 4); return f;
}
__device__ __forceinline__ unsigned short f2bf(float f) {
  unsigned int x; __builtin_memcpy(&x, &f, 4);
  unsigned int r = x + 0x7FFFu + ((x >> 16) & 1u);   // round-to-nearest-even
  return (unsigned short)(r >> 16);
}

// ---------------------------------------------------------------------------
// Weight pre-pack: f32 [Cout][Cin][K] -> bf16 in WMMA A-fragment order.
// Packed index: (((coT*(Cin/32)+ciC)*K + k)*32 + lane)*16 + h
//   hi=lane>>4, m=lane&15, j=h>>1, q=h&1,
//   ci = ciC*32 + (hi<<4) + h,  co = coT*16 + m
// ---------------------------------------------------------------------------
__global__ void pack_w(const float* __restrict__ w, unsigned short* __restrict__ wp,
                       int Cout, int Cin, int K)
{
  int i = blockIdx.x * blockDim.x + threadIdx.x;
  int total = Cout * Cin * K;
  if (i >= total) return;
  int h    = i & 15;
  int lane = (i >> 4) & 31;
  int rest = i >> 9;
  int k    = rest % K;   rest /= K;
  int nCi  = Cin >> 5;
  int ciC  = rest % nCi;
  int coT  = rest / nCi;
  int hi = lane >> 4, m = lane & 15;
  int j  = h >> 1,    q = h & 1;
  int ci = (ciC << 5) + (hi << 4) + h;
  int co = (coT << 4) + m;
  wp[i] = f2bf(w[((size_t)co * Cin + ci) * K + k]);
}

// ---------------------------------------------------------------------------
// 1-D conv as WMMA GEMM, K compile-time, 16(co) x 32(t) tile per wave
// (two accumulators share each A fragment).  LDS tile stored transposed
// [t][ci(32)] so a lane's B fragment is one contiguous 32B-aligned read.
// Staging keeps raw bf16 bits: ReLU is a sign-bit select, padding writes 0.
// ---------------------------------------------------------------------------
#define TSX 66   // >= 31*stride + (K-1)*dil + 1 for every layer here

template <int KT>
__global__ void __launch_bounds__(256) conv1d_wmma(
    const unsigned short* __restrict__ x,
    const unsigned short* __restrict__ wpk,
    const float* __restrict__ bias,
    unsigned short* __restrict__ y,
    const unsigned short* __restrict__ res,
    float* __restrict__ outT,
    int Bn, int Cin, int Cout, int Tin, int Tout,
    int stride, int pad, int dil,
    int relu_in, int relu_out)
{
  __shared__ __align__(64) unsigned short sx[TSX * 32];   // [t][ci]

  const int tid   = threadIdx.x;
  const int lane  = tid & 31;
  const int wave  = tid >> 5;
  const int bs    = blockDim.x;
  const int nw    = bs >> 5;
  const int b     = blockIdx.z;
  const int t0    = blockIdx.x * 32;
  const int coT   = blockIdx.y * nw + wave;
  const int co0   = coT * 16;
  const int tbase = t0 * stride - pad;
  const int nCi   = Cin >> 5;

  const int n  = lane & 15;
  const int hi = lane >> 4;

  // staging mapping: each thread owns one channel row, strided columns
  const int tpr  = nw;            // threads per LDS row = bs/32
  const int lrow = tid % tpr;
  const int rr   = tid / tpr;     // channel row 0..31

  v8f acc0 = {0.f, 0.f, 0.f, 0.f, 0.f, 0.f, 0.f, 0.f};
  v8f acc1 = acc0;

  for (int ciC = 0; ciC < nCi; ++ciC) {
    const int ci0 = ciC << 5;
    __syncthreads();                         // previous readers done
    {
      const unsigned short* xr = x + ((size_t)b * Cin + (ci0 + rr)) * Tin;
      for (int cc = lrow; cc < TSX; cc += tpr) {
        int tin = tbase + cc;
        unsigned short uv = 0;
        if (tin >= 0 && tin < Tin) uv = xr[tin];
        if (relu_in && (uv & 0x8000u)) uv = 0;   // bf16 ReLU = sign test
        sx[cc * 32 + rr] = uv;                   // transposed store
      }
    }
    __syncthreads();

    // ---- A fragments for all K taps: contiguous 32B/lane each ----
    const v16u* abase = (const v16u*)wpk + ((size_t)(coT * nCi + ciC) * KT) * 32 + lane;
    if (ciC + 1 < nCi)
      __builtin_prefetch((const void*)(abase + KT * 32), 0, 1);
    V16Cast a[KT];
    #pragma unroll
    for (int k = 0; k < KT; ++k) a[k].u = abase[k * 32];

    #pragma unroll
    for (int k = 0; k < KT; ++k) {
      V16Cast bm0, bm1;
      bm0.u = *(const v16u*)&sx[(n * stride + k * dil) * 32 + (hi << 4)];
      bm1.u = *(const v16u*)&sx[((n + 16) * stride + k * dil) * 32 + (hi << 4)];
      acc0 = __builtin_amdgcn_wmma_f32_16x16x32_bf16(
          false, a[k].b, false, bm0.b, (short)0, acc0, false, false);
      acc1 = __builtin_amdgcn_wmma_f32_16x16x32_bf16(
          false, a[k].b, false, bm1.b, (short)0, acc1, false, false);
    }
  }

  // ---- epilogue (C/D layout: row m = j + 8*hi, col = n / n+16) ----
  #pragma unroll
  for (int j = 0; j < 8; ++j) {
    int m  = j + (hi << 3);
    int co = co0 + m;
    float bv = bias[co];
    #pragma unroll
    for (int half = 0; half < 2; ++half) {
      int   t = t0 + n + half * 16;
      float v = (half ? acc1[j] : acc0[j]) + bv;
      if (res)  v += bf2f(res[((size_t)b * Cout + co) * Tout + t]);
      if (relu_out) v = fmaxf(v, 0.f);
      if (y)    y[((size_t)b * Cout + co) * Tout + t] = f2bf(v);
      if (outT) outT[((size_t)b * Tout + t) * Cout + co] = v;
    }
  }
}

// ---------------------------------------------------------------------------
// features f32 [B][T][C] -> bf16 [B][C][T]
// ---------------------------------------------------------------------------
__global__ void feat_to_nch(const float* __restrict__ f,
                            unsigned short* __restrict__ x,
                            int Bn, int T, int C)
{
  int i = blockIdx.x * blockDim.x + threadIdx.x;
  int total = Bn * T * C;
  if (i >= total) return;
  int c = i % C;
  int t = (i / C) % T;
  int b = i / (C * T);
  x[((size_t)b * C + c) * T + t] = f2bf(f[i]);
}

// ---------------------------------------------------------------------------
// nearest-neighbor upsample x2 along T (bf16)
// ---------------------------------------------------------------------------
__global__ void upsample2(const unsigned short* __restrict__ x,
                          unsigned short* __restrict__ y,
                          int Bn, int C, int Tin)
{
  int i = blockIdx.x * blockDim.x + threadIdx.x;
  int T2 = Tin * 2;
  int total = Bn * C * T2;
  if (i >= total) return;
  int t  = i % T2;
  int bc = i / T2;
  y[i] = x[(size_t)bc * Tin + (t >> 1)];
}

__global__ void zero_f(float* p, int n)
{
  int i = blockIdx.x * blockDim.x + threadIdx.x;
  if (i < n) p[i] = 0.f;
}

// ---------------------------------------------------------------------------
// VQ assignment: block per vector v=(b*T+t); argmin_code (c.c - 2 x.c).
// Accumulates commit sum and code counts via f32 atomics.
// ---------------------------------------------------------------------------
__global__ void __launch_bounds__(256) vq_assign(
    const unsigned short* __restrict__ xenc,  // [B][C][T] bf16
    const float* __restrict__ cb,             // [NC][C]
    int C, int T, int NC,
    int* __restrict__ idx,
    float* __restrict__ counts,
    float* __restrict__ commit)
{
  __shared__ float xf[512];
  __shared__ float sd[256];
  __shared__ int   si[256];

  int v = blockIdx.x;
  int b = v / T, t = v % T;
  int tid = threadIdx.x;

  for (int c = tid; c < C; c += 256)
    xf[c] = bf2f(xenc[((size_t)b * C + c) * T + t]);
  __syncthreads();

  float best = 3.4e38f; int bi = 0;
  for (int code = tid; code < NC; code += 256) {
    const float* cp = cb + (size_t)code * C;
    float dot = 0.f, cc = 0.f;
    for (int c = 0; c < C; ++c) { float cv = cp[c]; dot += xf[c] * cv; cc += cv * cv; }
    float d = cc - 2.f * dot;
    if (d < best) { best = d; bi = code; }
  }
  sd[tid] = best; si[tid] = bi;
  __syncthreads();
  for (int s = 128; s > 0; s >>= 1) {
    if (tid < s && sd[tid + s] < sd[tid]) { sd[tid] = sd[tid + s]; si[tid] = si[tid + s]; }
    __syncthreads();
  }
  int bestIdx = si[0];
  if (tid == 0) { idx[v] = bestIdx; atomicAdd(counts + bestIdx, 1.f); }
  __syncthreads();

  const float* cp = cb + (size_t)bestIdx * C;
  float part = 0.f;
  for (int c = tid; c < C; c += 256) { float df = xf[c] - cp[c]; part += df * df; }
  sd[tid] = part;
  __syncthreads();
  for (int s = 128; s > 0; s >>= 1) {
    if (tid < s) sd[tid] += sd[tid + s];
    __syncthreads();
  }
  if (tid == 0) atomicAdd(commit, sd[0]);
}

// ---------------------------------------------------------------------------
// Rotation trick (reference reduces over the T axis of [B][C][T]).
// One block per (b,c) row, blockDim == T == 128.
// ---------------------------------------------------------------------------
__global__ void __launch_bounds__(128) vq_rotate(
    const unsigned short* __restrict__ xenc,
    const float* __restrict__ cb,
    const int* __restrict__ idx,
    unsigned short* __restrict__ out,
    int C, int T)
{
  __shared__ float s0[128], s1[128], s2[128];
  int bc = blockIdx.x;
  int b = bc / C, c = bc % C;
  int t = threadIdx.x;

  float e = bf2f(xenc[((size_t)b * C + c) * T + t]);
  float q = cb[(size_t)idx[b * T + t] * C + c];

  s0[t] = e * e; s1[t] = q * q; s2[t] = e * q;
  __syncthreads();
  for (int s = 64; s > 0; s >>= 1) {
    if (t < s) { s0[t] += s0[t + s]; s1[t] += s1[t + s]; s2[t] += s2[t + s]; }
    __syncthreads();
  }
  float ne = fmaxf(sqrtf(s0[0]), 1e-12f);
  float nq = fmaxf(sqrtf(s1[0]), 1e-12f);
  float E  = s2[0];
  float nr = fmaxf(sqrtf(fmaxf(2.f + 2.f * E / (ne * nq), 0.f)), 1e-12f);
  float S1 = (ne + E / nq) / nr;        // sum_t r*e
  float en = e / ne, qn = q / nq;
  float r  = (en + qn) / nr;
  float Rx = e - 2.f * r * S1 + 2.f * qn * ne;   // S2 = sum e_n*e = ne
  out[((size_t)b * C + c) * T + t] = f2bf((nq / ne) * Rx);
}

// ---------------------------------------------------------------------------
// Finalize: commit mean + perplexity appended after the main output.
// ---------------------------------------------------------------------------
__global__ void __launch_bounds__(512) vq_finalize(
    const float* __restrict__ counts, const float* __restrict__ commit,
    float* __restrict__ out, int NV, int CD)
{
  __shared__ float s[512];
  int t = threadIdx.x;
  float p = counts[t] / (float)NV;
  s[t] = p * logf(p + 1e-7f);
  __syncthreads();
  for (int k = 256; k > 0; k >>= 1) {
    if (t < k) s[t] += s[t + k];
    __syncthreads();
  }
  if (t == 0) {
    out[0] = commit[0] / ((float)NV * (float)CD);  // commit
    out[1] = expf(-s[0]);                          // perplexity
  }
}

// ---------------------------------------------------------------------------
// Host side
// ---------------------------------------------------------------------------
struct CW { const float* w; const float* b; };

extern "C" void kernel_launch(void* const* d_in, const int* in_sizes, int n_in,
                              void* d_out, int out_size, void* d_ws, size_t ws_size,
                              hipStream_t stream)
{
  (void)in_sizes; (void)n_in; (void)out_size; (void)ws_size;

  const int Bn = 8, T0 = 1024, NF = 64, W = 512, NC = 512, CD = 512;

  // ---- walk inputs in setup_inputs() insertion order ----
  int p = 0;
  const float* features = (const float*)d_in[p++];
  auto cw = [&]() { CW c; c.w = (const float*)d_in[p++]; c.b = (const float*)d_in[p++]; return c; };

  CW enc_in = cw();
  CW enc_down[3]; CW enc_c1[3][3]; CW enc_c2[3][3];
  for (int d = 0; d < 3; ++d) {
    enc_down[d] = cw();
    for (int i = 0; i < 3; ++i) { enc_c1[d][i] = cw(); enc_c2[d][i] = cw(); }
  }
  CW enc_out = cw();
  CW dec_in = cw();
  CW dec_c1[3][3]; CW dec_c2[3][3]; CW dec_up[3];
  for (int u = 0; u < 3; ++u) {
    for (int i = 0; i < 3; ++i) { dec_c1[u][i] = cw(); dec_c2[u][i] = cw(); }
    dec_up[u] = cw();
  }
  CW dec_mid = cw();
  CW dec_out = cw();
  const float* codebook = (const float*)d_in[p++];

  // ---- workspace layout ----
  const size_t bufElems = (size_t)Bn * W * T0;             // 4M bf16 elements
  unsigned char* ws = (unsigned char*)d_ws;
  unsigned short* buf[3] = {
    (unsigned short*)(ws),
    (unsigned short*)(ws + bufElems * 2),
    (unsigned short*)(ws + bufElems * 4) };
  int*            idx    = (int*)  (ws + bufElems * 6);
  float*          counts = (float*)(ws + bufElems * 6 + 4096);
  float*          commit = counts + 512;
  unsigned short* wpack  = (unsigned short*)(ws + bufElems * 6 + 16384); // 2MB max

  // ---- conv launcher: pack weights, then WMMA conv ----
  auto conv = [&](const unsigned short* xin, int Cin, CW prm,
                  unsigned short* yout, int Cout, int Tin, int Tout,
                  int K, int stride, int pad, int dil,
                  int rin, int rout,
                  const unsigned short* res, float* outT) {
    int nwt = Cout * Cin * K;
    hipLaunchKernelGGL(pack_w, dim3((nwt + 255) / 256), dim3(256), 0, stream,
                       prm.w, wpack, Cout, Cin, K);
    int nw = (Cout >= 128) ? 8 : (Cout / 16);
    dim3 g(Tout / 32, Cout / (16 * nw), Bn);
    dim3 blk(32 * nw);
    if (K == 1)
      hipLaunchKernelGGL(conv1d_wmma<1>, g, blk, 0, stream, xin, wpack, prm.b, yout,
                         res, outT, Bn, Cin, Cout, Tin, Tout, stride, pad, dil, rin, rout);
    else if (K == 3)
      hipLaunchKernelGGL(conv1d_wmma<3>, g, blk, 0, stream, xin, wpack, prm.b, yout,
                         res, outT, Bn, Cin, Cout, Tin, Tout, stride, pad, dil, rin, rout);
    else
      hipLaunchKernelGGL(conv1d_wmma<4>, g, blk, 0, stream, xin, wpack, prm.b, yout,
                         res, outT, Bn, Cin, Cout, Tin, Tout, stride, pad, dil, rin, rout);
  };
  const int dils[3] = {1, 3, 9};

  // ================= Encoder =================
  {
    int n = Bn * T0 * NF;
    hipLaunchKernelGGL(feat_to_nch, dim3((n + 255) / 256), dim3(256), 0, stream,
                       features, buf[0], Bn, T0, NF);
  }
  conv(buf[0], NF, enc_in, buf[1], W, T0, T0, 3, 1, 1, 1, 0, 1, nullptr, nullptr);
  int cur = 1, Tc = T0;
  for (int d = 0; d < 3; ++d) {
    int nTc = Tc / 2;
    int nx = (cur + 1) % 3;
    conv(buf[cur], W, enc_down[d], buf[nx], W, Tc, nTc, 4, 2, 1, 1, 0, 0, nullptr, nullptr);
    cur = nx; Tc = nTc;
    for (int i = 0; i < 3; ++i) {
      int dl = dils[i];
      int h = (cur + 1) % 3, o = (cur + 2) % 3;
      conv(buf[cur], W, enc_c1[d][i], buf[h], W, Tc, Tc, 3, 1, dl, dl, 1, 0, nullptr, nullptr);
      conv(buf[h],   W, enc_c2[d][i], buf[o], W, Tc, Tc, 1, 1, 0, 1, 1, 0, buf[cur], nullptr);
      cur = o;
    }
  }
  int eo = (cur + 1) % 3;
  conv(buf[cur], W, enc_out, buf[eo], W, Tc, Tc, 3, 1, 1, 1, 0, 0, nullptr, nullptr);
  // Tc == 128; x_enc in buf[eo]

  // ================= Quantizer + rotation trick =================
  hipLaunchKernelGGL(zero_f, dim3(3), dim3(256), 0, stream, counts, NC + 1);
  hipLaunchKernelGGL(vq_assign, dim3(Bn * Tc), dim3(256), 0, stream,
                     buf[eo], codebook, W, Tc, NC, idx, counts, commit);
  int qb = (eo + 1) % 3;
  hipLaunchKernelGGL(vq_rotate, dim3(Bn * W), dim3(128), 0, stream,
                     buf[eo], codebook, idx, buf[qb], W, Tc);

  // ================= Decoder =================
  cur = qb;
  int nx = (cur + 1) % 3;
  conv(buf[cur], W, dec_in, buf[nx], W, Tc, Tc, 3, 1, 1, 1, 0, 1, nullptr, nullptr);
  cur = nx;
  for (int u = 0; u < 3; ++u) {
    for (int i = 0; i < 3; ++i) {
      int dl = dils[2 - i];                          // reversed dilations
      int h = (cur + 1) % 3, o = (cur + 2) % 3;
      conv(buf[cur], W, dec_c1[u][i], buf[h], W, Tc, Tc, 3, 1, dl, dl, 1, 0, nullptr, nullptr);
      conv(buf[h],   W, dec_c2[u][i], buf[o], W, Tc, Tc, 1, 1, 0, 1, 1, 0, buf[cur], nullptr);
      cur = o;
    }
    nx = (cur + 1) % 3;
    {
      int n = Bn * W * Tc * 2;
      hipLaunchKernelGGL(upsample2, dim3((n + 255) / 256), dim3(256), 0, stream,
                         buf[cur], buf[nx], Bn, W, Tc);
    }
    Tc *= 2; cur = nx;
    nx = (cur + 1) % 3;
    conv(buf[cur], W, dec_up[u], buf[nx], W, Tc, Tc, 3, 1, 1, 1, 0, 0, nullptr, nullptr);
    cur = nx;
  }
  nx = (cur + 1) % 3;
  conv(buf[cur], W, dec_mid, buf[nx], W, Tc, Tc, 3, 1, 1, 1, 0, 1, nullptr, nullptr);
  cur = nx;
  conv(buf[cur], W, dec_out, nullptr, NF, Tc, Tc, 3, 1, 1, 1, 0, 0, nullptr, (float*)d_out);

  hipLaunchKernelGGL(vq_finalize, dim3(1), dim3(512), 0, stream,
                     counts, commit, (float*)d_out + (size_t)Bn * T0 * NF,
                     Bn * 128, CD);
}
